// Decoder_83751862272585
// MI455X (gfx1250) — matrix operand order repaired and checked by reading, not compiled, not run on hardware
//
#include <hip/hip_runtime.h>

// ---------------------------------------------------------------------------
// Problem constants (match reference)
// S_LEN=128, B=128, H=1024, V=32000, L=2, H2=512, WSZ=10, MAXW=21
// ---------------------------------------------------------------------------
#define Bsz   128
#define Hdim  1024
#define Vdim  32000
#define H2dim 512
#define MAXW  21
#define KC    1024            // K chunk staged in LDS
#define RS    (KC + 8)        // LDS row stride (ushorts): bank-conflict padding

typedef __attribute__((ext_vector_type(16))) unsigned short v16us;
typedef __attribute__((ext_vector_type(8)))  unsigned short us8;
typedef __attribute__((ext_vector_type(8)))  unsigned int   v8ui;
typedef __attribute__((ext_vector_type(16))) __bf16         v16bf;
typedef __attribute__((ext_vector_type(8)))  float          v8f;

// round-to-nearest-even f32 -> bf16 (bit pattern)
__device__ __forceinline__ unsigned short f2bf(float f) {
    unsigned u = __float_as_uint(f);
    return (unsigned short)((u + 0x7FFFu + ((u >> 16) & 1u)) >> 16);
}

// pack two f32 -> one dword holding two bf16 (lo, hi)
__device__ __forceinline__ unsigned cvt2(float lo, float hi) {
    return (unsigned)f2bf(lo) | ((unsigned)f2bf(hi) << 16);
}

// A-matrix fragment (16x32 bf16) from a *pre-converted* bf16 (ushort) buffer.
// ISA 7.12.2: lanes 0-15 row m=lane: elems 0..7 = K 0..7, 8..15 = K 16..23;
// lanes 16-31 row m=lane-16: elems 0..7 = K 8..15, 8..15 = K 24..31.
__device__ __forceinline__ v16bf load_a_bf(const unsigned short* __restrict__ A,
                                           int lda, int mBase, int kk, int lane) {
    int m  = mBase + (lane & 15);
    int k0 = kk + ((lane >> 4) << 3);
    const us8* p0 = reinterpret_cast<const us8*>(A + (size_t)m * lda + k0);
    const us8* p1 = reinterpret_cast<const us8*>(A + (size_t)m * lda + k0 + 16);
    us8 x0 = p0[0], x1 = p1[0];
    v16us u;
    #pragma unroll
    for (int i = 0; i < 8; ++i) { u[i] = x0[i]; u[8 + i] = x1[i]; }
    return __builtin_bit_cast(v16bf, u);
}

// B-matrix fragment (32x16 bf16) from the LDS-staged tile.
// Lane holds column n=lane&15; lanes 0-15: K=kk+0..7,16..23; lanes 16-31: kk+8..15,24..31.
__device__ __forceinline__ v16bf load_b_lds(const unsigned short* bt, int kk, int lane) {
    int n  = lane & 15;
    int k0 = kk + ((lane >> 4) << 3);
    const us8* p = reinterpret_cast<const us8*>(bt + n * RS + k0);
    us8 x0 = p[0], x1 = p[2];
    v16us u;
    #pragma unroll
    for (int i = 0; i < 8; ++i) { u[i] = x0[i]; u[8 + i] = x1[i]; }
    return __builtin_bit_cast(v16bf, u);
}

// ---------------------------------------------------------------------------
// Unified WMMA GEMM: D(128 x N) = A(128 x KT) @ W(N x KT)^T, per-mode epilogue
//   MODE 0: gates  -> out0 = D + bias0 + bias1          (W spliced Wih|Whh)
//   MODE 1: fc1    -> out0 = tanh(D + bias0), out1 = bf16(out0)
//   MODE 2: fc2    -> out0 = D + bias0
// grid.x = N/16 tiles; 8 waves/block = 8 M-tiles (M=128).
// Weight tile converted to bf16 in LDS once per block per K-chunk.
// ---------------------------------------------------------------------------
template<int KT, int MODE>
__global__ void gemm_wmma_kernel(const unsigned short* __restrict__ A, int lda,
                                 const float* __restrict__ W0,
                                 const float* __restrict__ W1, int ldw,
                                 const float* __restrict__ bias0,
                                 const float* __restrict__ bias1,
                                 float* __restrict__ out0,
                                 unsigned short* __restrict__ out1, int ldo) {
    __shared__ unsigned short btile[16 * RS];     // 33 KB
    unsigned* lds32 = reinterpret_cast<unsigned*>(btile);
    int tid   = threadIdx.x;
    int lane  = tid & 31;
    int mBase = (tid >> 5) * 16;
    int nBase = blockIdx.x * 16;
    v8f c = {};
    for (int kh = 0; kh < KT; kh += KC) {
        // ---- stage 16 x KC weight chunk into LDS as bf16 (converted once) ----
        const float* src = (MODE == 0 && kh > 0) ? W1 : W0;
        int kglb = (MODE == 0) ? 0 : kh;          // gates: W1 restarts at k=0
        for (int i = tid; i < 16 * KC / 4; i += 256) {
            int e = i * 4;
            int n = e >> 10;                      // e / KC
            int k = e & (KC - 1);                 // e % KC
            const float4* wp = reinterpret_cast<const float4*>(
                src + (size_t)(nBase + n) * ldw + kglb + k);
            float4 w = wp[0];
            unsigned* dst = lds32 + ((n * RS + k) >> 1);
            dst[0] = cvt2(w.x, w.y);
            dst[1] = cvt2(w.z, w.w);
        }
        __syncthreads();
        // ---- WMMA over this chunk ----
        for (int kk = 0; kk < KC; kk += 32) {
            v16bf a = load_a_bf(A, lda, mBase, kh + kk, lane);
            v16bf b = load_b_lds(btile, kk, lane);
            c = __builtin_amdgcn_wmma_f32_16x16x32_bf16(false, a, false, b,
                                                        (short)0, c, false, false);
        }
        __syncthreads();
    }
    int col     = nBase + (lane & 15);
    int rowBase = mBase + ((lane >> 4) << 3);
    if (MODE == 0) {
        float bsum = bias0[col] + bias1[col];
        #pragma unroll
        for (int j = 0; j < 8; ++j)
            out0[(size_t)(rowBase + j) * ldo + col] = c[j] + bsum;
    } else if (MODE == 1) {
        float bv = bias0[col];
        #pragma unroll
        for (int j = 0; j < 8; ++j) {
            float v = tanhf(c[j] + bv);
            out0[(size_t)(rowBase + j) * ldo + col] = v;
            out1[(size_t)(rowBase + j) * ldo + col] = f2bf(v);
        }
    } else {
        float bv = bias0[col];
        #pragma unroll
        for (int j = 0; j < 8; ++j)
            out0[(size_t)(rowBase + j) * ldo + col] = c[j] + bv;
    }
}

// ---------------------------------------------------------------------------
// K1: xcat_bf = [embedding[word] | h0[0]] as bf16   (B x 2048)
// ---------------------------------------------------------------------------
__global__ void embed_cat_kernel(const float* __restrict__ emb,
                                 const int* __restrict__ word,
                                 const float* __restrict__ h0,
                                 unsigned short* __restrict__ xcat_bf) {
    int b = blockIdx.x, tid = threadIdx.x;
    int w = word[b];
    #pragma unroll
    for (int t = 0; t < 4; ++t) {
        int h = tid + t * 256;
        xcat_bf[(size_t)b * 2048 + h]        = f2bf(emb[(size_t)w * Hdim + h]);
        xcat_bf[(size_t)b * 2048 + 1024 + h] = f2bf(h0[(size_t)b * Hdim + h]);
    }
}

// ---------------------------------------------------------------------------
// K3: LSTM cell elementwise; also stage next GEMM input (bf16) / attn input
// ---------------------------------------------------------------------------
__global__ void lstm_cell_kernel(const float* __restrict__ gates,
                                 const float* __restrict__ c0l,
                                 float* __restrict__ hOut,
                                 float* __restrict__ cOut,
                                 unsigned short* __restrict__ bfdst, // xcat_bf or null
                                 const float* __restrict__ h0next,   // with bfdst
                                 float* __restrict__ f32dst) {       // outx or null
    int b = blockIdx.x, tid = threadIdx.x;
    #pragma unroll
    for (int t = 0; t < 4; ++t) {
        int j = tid + t * 256;
        float gi = gates[(size_t)b * 4096 + j];
        float gf = gates[(size_t)b * 4096 + 1024 + j];
        float gg = gates[(size_t)b * 4096 + 2048 + j];
        float go = gates[(size_t)b * 4096 + 3072 + j];
        float si = 1.f / (1.f + expf(-gi));
        float sf = 1.f / (1.f + expf(-gf));
        float so = 1.f / (1.f + expf(-go));
        float cv = sf * c0l[(size_t)b * Hdim + j] + si * tanhf(gg);
        float hv = so * tanhf(cv);
        cOut[(size_t)b * Hdim + j] = cv;
        hOut[(size_t)b * Hdim + j] = hv;
        if (bfdst) {
            bfdst[(size_t)b * 2048 + j]        = f2bf(hv);
            bfdst[(size_t)b * 2048 + 1024 + j] = f2bf(h0next[(size_t)b * Hdim + j]);
        }
        if (f32dst) f32dst[(size_t)b * Hdim + j] = hv;
    }
}

// ---------------------------------------------------------------------------
// K4: p[b] = S * sigmoid(tanh(out[b] @ w1^T + b1) @ w2^T + b2)
// ---------------------------------------------------------------------------
__global__ void attn_p_kernel(const float* __restrict__ outx,
                              const float* __restrict__ w1,
                              const float* __restrict__ b1,
                              const float* __restrict__ w2,
                              const float* __restrict__ b2,
                              const int* __restrict__ Sptr,
                              float* __restrict__ p) {
    __shared__ float xsh[Hdim];
    __shared__ float red[H2dim];
    int b = blockIdx.x, tid = threadIdx.x;   // 512 threads
    xsh[tid]       = outx[(size_t)b * Hdim + tid];
    xsh[tid + 512] = outx[(size_t)b * Hdim + tid + 512];
    __syncthreads();
    float acc = b1[tid];
    const float* wrow = w1 + (size_t)tid * Hdim;
    for (int k = 0; k < Hdim; ++k) acc += xsh[k] * wrow[k];
    red[tid] = tanhf(acc) * w2[tid];
    __syncthreads();
    for (int s = 256; s > 0; s >>= 1) {
        if (tid < s) red[tid] += red[tid + s];
        __syncthreads();
    }
    if (tid == 0) {
        float Sf = (float)(*Sptr);
        p[b] = Sf / (1.f + expf(-(red[0] + b2[0])));
    }
}

// ---------------------------------------------------------------------------
// K5: local-attention window: scores, softmax*gauss -> a, ctx; build cat (bf16)
// ---------------------------------------------------------------------------
__global__ void attn_ctx_kernel(const float* __restrict__ outx,
                                const float* __restrict__ enc,
                                const float* __restrict__ p,
                                const int* __restrict__ Sptr,
                                unsigned short* __restrict__ xcat_bf,
                                float* __restrict__ aOut) {
    __shared__ float xsh[Hdim];
    __shared__ float red[256];
    __shared__ float sa[MAXW];
    int b = blockIdx.x, tid = threadIdx.x;
    int S = *Sptr;
    #pragma unroll
    for (int t = 0; t < 4; ++t) xsh[tid + t * 256] = outx[(size_t)b * Hdim + tid + t * 256];
    float pb = p[b];
    int wsi = (int)rintf(fmaxf(pb - 10.f, 0.f));
    int wei = (int)rintf(fminf(pb + 10.f, (float)(S - 1)));
    __syncthreads();
    for (int w = 0; w < MAXW; ++w) {
        int idx  = wsi + w;
        int cidx = idx < S - 1 ? idx : S - 1;
        float acc = 0.f;
        const float* ep = enc + ((size_t)cidx * Bsz + b) * Hdim;
        for (int h = tid; h < Hdim; h += 256) acc += xsh[h] * ep[h];
        red[tid] = acc;
        __syncthreads();
        for (int s = 128; s > 0; s >>= 1) {
            if (tid < s) red[tid] += red[tid + s];
            __syncthreads();
        }
        if (tid == 0) sa[w] = (idx <= wei) ? red[0] : 0.f;
        __syncthreads();
    }
    if (tid == 0) {
        float mx = sa[0];
        for (int w = 1; w < MAXW; ++w) mx = fmaxf(mx, sa[w]);
        float e[MAXW], sum = 0.f;
        for (int w = 0; w < MAXW; ++w) { e[w] = expf(sa[w] - mx); sum += e[w]; }
        for (int w = 0; w < MAXW; ++w) {
            int idx = wsi + w;
            float g = (idx <= wei) ? expf(((float)idx - pb) / 50.f) : 0.f; // 2*std2 = 50
            float av = e[w] / sum * g;
            sa[w] = av;
            aOut[(size_t)b * MAXW + w] = av;
        }
    }
    __syncthreads();
    for (int h = tid; h < Hdim; h += 256) {
        float acc = 0.f;
        for (int w = 0; w < MAXW; ++w) {
            int idx = wsi + w;
            if (idx <= wei) {
                int cidx = idx < S - 1 ? idx : S - 1;
                acc += sa[w] * enc[((size_t)cidx * Bsz + b) * Hdim + h];
            }
        }
        xcat_bf[(size_t)b * 2048 + h]        = f2bf(acc);     // ctx
        xcat_bf[(size_t)b * 2048 + 1024 + h] = f2bf(xsh[h]);  // out
    }
}

// ---------------------------------------------------------------------------
// K8: y = log_softmax(logits) per row of 32000
// ---------------------------------------------------------------------------
__global__ void logsoftmax_kernel(const float* __restrict__ logits,
                                  float* __restrict__ y) {
    __shared__ float red[256];
    int b = blockIdx.x, tid = threadIdx.x;
    const float* row = logits + (size_t)b * Vdim;
    float mx = -3.402823466e38f;
    for (int i = tid; i < Vdim; i += 256) mx = fmaxf(mx, row[i]);
    red[tid] = mx;
    __syncthreads();
    for (int s = 128; s > 0; s >>= 1) {
        if (tid < s) red[tid] = fmaxf(red[tid], red[tid + s]);
        __syncthreads();
    }
    mx = red[0];
    __syncthreads();
    float sum = 0.f;
    for (int i = tid; i < Vdim; i += 256) sum += expf(row[i] - mx);
    red[tid] = sum;
    __syncthreads();
    for (int s = 128; s > 0; s >>= 1) {
        if (tid < s) red[tid] += red[tid + s];
        __syncthreads();
    }
    float lse = mx + logf(red[0]);
    float* yr = y + (size_t)b * Vdim;
    for (int i = tid; i < Vdim; i += 256) yr[i] = row[i] - lse;
}

// ---------------------------------------------------------------------------
// Host launch
// ---------------------------------------------------------------------------
extern "C" void kernel_launch(void* const* d_in, const int* in_sizes, int n_in,
                              void* d_out, int out_size, void* d_ws, size_t ws_size,
                              hipStream_t stream) {
    (void)in_sizes; (void)n_in; (void)out_size; (void)ws_size;
    const int*   Sptr = (const int*)d_in[0];
    const float* enc  = (const float*)d_in[1];
    const int*   word = (const int*)d_in[2];
    const float* h0   = (const float*)d_in[3];
    const float* c0   = (const float*)d_in[4];
    const float* emb  = (const float*)d_in[5];
    const float* Wih  = (const float*)d_in[6];
    const float* Whh  = (const float*)d_in[7];
    const float* bih  = (const float*)d_in[8];
    const float* bhh  = (const float*)d_in[9];
    const float* aw1  = (const float*)d_in[10];
    const float* ab1  = (const float*)d_in[11];
    const float* aw2  = (const float*)d_in[12];
    const float* ab2  = (const float*)d_in[13];
    const float* f1w  = (const float*)d_in[14];
    const float* f1b  = (const float*)d_in[15];
    const float* f2w  = (const float*)d_in[16];
    const float* f2b  = (const float*)d_in[17];

    float* out = (float*)d_out;
    float* ws  = (float*)d_ws;

    // workspace layout (float units)
    unsigned short* xcat_bf = (unsigned short*)ws;          // 128*2048 bf16 = 131072 f
    float* gates  = ws + 131072;                            // 128*4096 = 524288
    float* outx   = ws + 655360;                            // 128*1024 = 131072
    float* pbuf   = ws + 786432;                            // 128
    unsigned short* o2bf = (unsigned short*)(ws + 786560);  // 128*1024 bf16 = 65536 f
    float* logits = ws + 852096;                            // 128*32000 = 4096000
    // total: 4,948,096 floats (~19.8 MB)

    // output layout (floats), return order: y, o2, h_new, c_new, a
    float* y  = out;                   // 4096000
    float* o2 = out + 4096000;         // 131072
    float* hN = out + 4227072;         // 262144
    float* cN = out + 4489216;         // 262144
    float* aO = out + 4751360;         // 2688

    const size_t Wl = 4096u * 1024u;      // per-layer weight stride
    const size_t bl = 4096u;              // per-layer bias stride
    const size_t hl = (size_t)Bsz * Hdim; // per-layer h/c stride

    embed_cat_kernel<<<Bsz, 256, 0, stream>>>(emb, word, h0, xcat_bf);

    // layer 0: gates = [x|h] @ [Wih|Whh]^T + biases
    gemm_wmma_kernel<2048, 0><<<256, 256, 0, stream>>>(
        xcat_bf, 2048, Wih, Whh, 1024, bih, bhh, gates, nullptr, 4096);
    lstm_cell_kernel<<<Bsz, 256, 0, stream>>>(gates, c0, hN, cN,
                                              xcat_bf, h0 + hl, nullptr);
    // layer 1
    gemm_wmma_kernel<2048, 0><<<256, 256, 0, stream>>>(
        xcat_bf, 2048, Wih + Wl, Whh + Wl, 1024, bih + bl, bhh + bl,
        gates, nullptr, 4096);
    lstm_cell_kernel<<<Bsz, 256, 0, stream>>>(gates, c0 + hl, hN + hl, cN + hl,
                                              nullptr, nullptr, outx);

    attn_p_kernel<<<Bsz, 512, 0, stream>>>(outx, aw1, ab1, aw2, ab2, Sptr, pbuf);
    attn_ctx_kernel<<<Bsz, 256, 0, stream>>>(outx, enc, pbuf, Sptr, xcat_bf, aO);

    // fc1: o2 = tanh(cat @ fc1_w^T + b)
    gemm_wmma_kernel<2048, 1><<<64, 256, 0, stream>>>(
        xcat_bf, 2048, f1w, nullptr, 2048, f1b, nullptr, o2, o2bf, Hdim);
    // fc2: logits = o2 @ fc2_w^T + b   (dominant GEMM)
    gemm_wmma_kernel<1024, 2><<<2000, 256, 0, stream>>>(
        o2bf, 1024, f2w, nullptr, 1024, f2b, nullptr, logits, nullptr, Vdim);

    logsoftmax_kernel<<<Bsz, 256, 0, stream>>>(logits, y);
}
